// MultiHeadSelfAttention_21706764714760
// MI455X (gfx1250) — matrix operand order repaired and checked by reading, not compiled, not run on hardware
//
#include <hip/hip_runtime.h>

// ---------------------------------------------------------------------------
// Problem constants (from the reference)
// ---------------------------------------------------------------------------
constexpr int D_IN   = 1024;
constexpr int D_OUT  = 1024;
constexpr int NH     = 16;
constexpr int DH     = 64;
constexpr int BATCH  = 2;
constexpr int SEQ    = 2048;
constexpr float SCORE_SCALE = 1.0f / (64.0f * 64.0f);   // reference: / HEAD_DIM**2

// WMMA types (CDNA5, wave32)
typedef __attribute__((ext_vector_type(16))) __bf16 v16bf;
typedef __attribute__((ext_vector_type(8)))  float  v8f;

union FragBF { v16bf v; uint4 q[2]; };

__device__ __forceinline__ v8f wmma_bf16(v16bf a, v16bf b, v8f c) {
    return __builtin_amdgcn_wmma_f32_16x16x32_bf16(false, a, false, b, (short)0, c,
                                                   false, false);
}

// float -> bf16 (RNE) bit helpers
__device__ __forceinline__ unsigned int bf16_bits(float x) {
    unsigned int u = __float_as_uint(x);
    u += 0x7FFFu + ((u >> 16) & 1u);
    return u >> 16;
}
__device__ __forceinline__ unsigned int bfpair(float lo, float hi) {
    unsigned int a = __float_as_uint(lo);
    unsigned int b = __float_as_uint(hi);
    a += 0x7FFFu + ((a >> 16) & 1u);
    b += 0x7FFFu + ((b >> 16) & 1u);
    return (a >> 16) | (b & 0xFFFF0000u);
}
__device__ __forceinline__ uint4 bfpack8(float4 f0, float4 f1) {
    uint4 u;
    u.x = bfpair(f0.x, f0.y);
    u.y = bfpair(f0.z, f0.w);
    u.z = bfpair(f1.x, f1.y);
    u.w = bfpair(f1.z, f1.w);
    return u;
}

// ---------------------------------------------------------------------------
// Async global->LDS 16-byte copy (CDNA5 ASYNCcnt path).
// Builtin param0 = int __vector(4) ptr, global AS; param1 = LDS AS; imm, imm.
// Watermark waits enable double-buffered copy/compute overlap (async loads
// complete in order per wave).
// ---------------------------------------------------------------------------
#if __has_builtin(__builtin_amdgcn_global_load_async_to_lds_b128)
typedef __attribute__((__vector_size__(4 * sizeof(int)))) int vsi4;
typedef __attribute__((address_space(1))) vsi4 GV4;
typedef __attribute__((address_space(3))) vsi4 LV4;
__device__ __forceinline__ void cp_async16(void* lds, const void* g) {
    __builtin_amdgcn_global_load_async_to_lds_b128((GV4*)g, (LV4*)lds, 0, 0);
}
template <int N> __device__ __forceinline__ void cp_async_wait() {
#if __has_builtin(__builtin_amdgcn_s_wait_asynccnt)
    __builtin_amdgcn_s_wait_asynccnt(N);
#else
    asm volatile("s_wait_asynccnt %0" :: "i"(N) : "memory");
#endif
}
#elif defined(__AMDGCN__)
__device__ __forceinline__ void cp_async16(void* lds, const void* g) {
    unsigned int laddr = (unsigned int)(unsigned long long)lds;
    asm volatile("global_load_async_to_lds_b128 %0, %1, off"
                 :: "v"(laddr), "v"(g) : "memory");
}
template <int N> __device__ __forceinline__ void cp_async_wait() {
    asm volatile("s_wait_asynccnt %0" :: "i"(N) : "memory");
}
#else
__device__ __forceinline__ void cp_async16(void* lds, const void* g) {
    *(uint4*)lds = *(const uint4*)g;
}
template <int N> __device__ __forceinline__ void cp_async_wait() {}
#endif

// A-operand fragment (16x32 bf16, ISA 16-bit A layout): two ds_load_b128
__device__ __forceinline__ v16bf load_fragA(const unsigned short* rowptr, int kb, int g) {
    FragBF f;
    f.q[0] = *(const uint4*)(rowptr + kb + 8 * g);
    f.q[1] = *(const uint4*)(rowptr + kb + 16 + 8 * g);
    return f.v;
}
// B-operand fragment from B^T rows: two 16-byte chunks (same k order as A)
__device__ __forceinline__ v16bf load_fragB(const unsigned short* rowTptr, int kb, int g) {
    FragBF f;
    f.q[0] = *(const uint4*)(rowTptr + kb + 8 * g);
    f.q[1] = *(const uint4*)(rowTptr + kb + 16 + 8 * g);
    return f.v;
}

// ---------------------------------------------------------------------------
// Kernel A: convert X (f32) -> bf16, flat. 8 elems/thread.
// ---------------------------------------------------------------------------
__global__ __launch_bounds__(256) void cvt_x_kernel(const float* __restrict__ X,
                                                    unsigned short* __restrict__ Xb) {
    size_t i = ((size_t)blockIdx.x * 256 + threadIdx.x) * 8;
    const float4* p = (const float4*)(X + i);
    float4 f0 = p[0], f1 = p[1];
    *(uint4*)(Xb + i) = bfpack8(f0, f1);
}

// ---------------------------------------------------------------------------
// Kernel B: convert W (f32 [k][n]) -> bf16 transposed Wt [n][k], 3 matrices.
// ---------------------------------------------------------------------------
__global__ __launch_bounds__(256) void cvt_wT_kernel(
    const float* __restrict__ Wq, const float* __restrict__ Wk,
    const float* __restrict__ Wv, unsigned short* __restrict__ WtAll) {
    const float* W = (blockIdx.z == 0) ? Wq : (blockIdx.z == 1) ? Wk : Wv;
    unsigned short* T = WtAll + (size_t)blockIdx.z * D_IN * D_OUT;

    __shared__ __align__(16) unsigned short tile[64][64 + 8];
    const int kb = blockIdx.x * 64, nb = blockIdx.y * 64;
    const int tr = threadIdx.x >> 2;
    const int tc = (threadIdx.x & 3) * 16;

    #pragma unroll
    for (int i = 0; i < 16; i += 8) {
        const float* gp = W + (size_t)(kb + tr) * D_OUT + nb + tc + i;
        float4 f0 = ((const float4*)gp)[0];
        float4 f1 = ((const float4*)gp)[1];
        uint4 u = bfpack8(f0, f1);
        unsigned short tmp[8];
        *(uint4*)tmp = u;
        #pragma unroll
        for (int e = 0; e < 8; ++e) tile[tc + i + e][tr] = tmp[e];
    }
    __syncthreads();
    #pragma unroll
    for (int i = 0; i < 16; i += 8)
        *(uint4*)(T + (size_t)(nb + tr) * D_IN + kb + tc + i) = *(uint4*)&tile[tr][tc + i];
}

// ---------------------------------------------------------------------------
// Kernel C: QKV projection, double-buffered async-to-LDS staging.
// grid = (32,16,3), block 256. 3 async instr per thread per k-tile.
// ---------------------------------------------------------------------------
__global__ __launch_bounds__(256) void qkv_proj_kernel(
    const unsigned short* __restrict__ Xb, const unsigned short* __restrict__ WtAll,
    const float* __restrict__ bq, const float* __restrict__ bk,
    const float* __restrict__ bv,
    unsigned short* __restrict__ qo, unsigned short* __restrict__ ko,
    unsigned short* __restrict__ vo) {
    const unsigned short* W = WtAll + (size_t)blockIdx.z * D_IN * D_OUT;
    const float* bias;  unsigned short* out;
    if (blockIdx.z == 0)      { bias = bq; out = qo; }
    else if (blockIdx.z == 1) { bias = bk; out = ko; }
    else                      { bias = bv; out = vo; }

    __shared__ __align__(16) unsigned short xs[2][128][32];
    __shared__ __align__(16) unsigned short wt[2][64][32];

    const int tid  = threadIdx.x;
    const int lane = tid & 31;
    const int wv   = tid >> 5;
    const int g    = lane >> 4;
    const int wm   = (wv >> 1) * 32;
    const int wn   = (wv & 1) * 32;
    const int gm0  = blockIdx.x * 128;
    const int nb   = blockIdx.y * 64;

    auto issue_tile = [&](int kb, int buf) {
        #pragma unroll
        for (int i = 0; i < 2; ++i) {
            int c = tid + 256 * i;
            int row = c >> 2, off = (c & 3) * 8;
            cp_async16(&xs[buf][row][off], Xb + (size_t)(gm0 + row) * D_IN + kb + off);
        }
        int n = tid >> 2, off = (tid & 3) * 8;
        cp_async16(&wt[buf][n][off], W + (size_t)(nb + n) * D_IN + kb + off);
    };

    v8f acc[2][2];
    #pragma unroll
    for (int i = 0; i < 2; ++i)
        #pragma unroll
        for (int j = 0; j < 2; ++j)
            #pragma unroll
            for (int r = 0; r < 8; ++r) acc[i][j][r] = 0.0f;

    constexpr int NT = D_IN / 32;
    issue_tile(0, 0);
    for (int it = 0; it < NT; ++it) {
        const int cur = it & 1;
        if (it + 1 < NT) {           // prefetch next tile, keep it in flight
            issue_tile((it + 1) * 32, cur ^ 1);
            cp_async_wait<3>();      // drain current tile only (in-order)
        } else {
            cp_async_wait<0>();
        }
        __syncthreads();

        #pragma unroll
        for (int mt = 0; mt < 2; ++mt) {
            v16bf a = load_fragA(&xs[cur][wm + mt * 16 + (lane & 15)][0], 0, g);
            #pragma unroll
            for (int nt = 0; nt < 2; ++nt) {
                v16bf b = load_fragB(&wt[cur][wn + nt * 16 + (lane & 15)][0], 0, g);
                acc[mt][nt] = wmma_bf16(a, b, acc[mt][nt]);
            }
        }
        __syncthreads();             // protect buffer before it is re-filled
    }

    // epilogue: bias add, bf16 store; V goes out transposed [B,H,Dh,S]
    #pragma unroll
    for (int mt = 0; mt < 2; ++mt) {
        #pragma unroll
        for (int nt = 0; nt < 2; ++nt) {
            int gn = nb + wn + nt * 16 + (lane & 15);
            float bcol = bias[gn];
            int h = gn >> 6, d = gn & 63;
            #pragma unroll
            for (int r = 0; r < 8; ++r) {
                int gm = gm0 + wm + mt * 16 + 8 * g + r;
                int b_ = gm >> 11, s = gm & 2047;
                float val = acc[mt][nt][r] + bcol;
                size_t bh = (size_t)b_ * NH + h;
                if (blockIdx.z == 2)
                    out[(bh * DH + d) * SEQ + s] = (unsigned short)bf16_bits(val);
                else
                    out[(bh * SEQ + s) * DH + d] = (unsigned short)bf16_bits(val);
            }
        }
    }
}

// ---------------------------------------------------------------------------
// Kernel D: flash-style attention, double-buffered K/V async staging.
// grid = B*NH*(SEQ/128), block 256 (8 waves x 16 Q rows).
// 4 async instr per thread per j-tile.
// ---------------------------------------------------------------------------
__global__ __launch_bounds__(256) void attn_kernel(
    const unsigned short* __restrict__ q, const unsigned short* __restrict__ k,
    const unsigned short* __restrict__ v, float* __restrict__ out) {
    const int gid = blockIdx.x;
    const int qt  = gid & 15;
    const int bh  = gid >> 4;
    const int b_  = bh >> 4, h = bh & 15;

    const unsigned short* Qp = q + (size_t)bh * SEQ * DH;
    const unsigned short* Kp = k + (size_t)bh * SEQ * DH;
    const unsigned short* Vp = v + (size_t)bh * DH * SEQ;   // pre-transposed [d][s]

    __shared__ __align__(16) unsigned short qs[128][DH];
    __shared__ __align__(16) unsigned short ks[2][64][DH];
    __shared__ __align__(16) unsigned short vt[2][DH][64];
    __shared__ __align__(16) unsigned short ps[8][16][64];

    const int tid  = threadIdx.x;
    const int lane = tid & 31;
    const int w    = tid >> 5;
    const int g    = lane >> 4;

    auto issue_kv = [&](int j, int buf) {
        #pragma unroll
        for (int i = 0; i < 2; ++i) {
            int c = tid + 256 * i;
            int row = c >> 3, off = (c & 7) * 8;
            cp_async16(&ks[buf][row][off], Kp + (size_t)(j * 64 + row) * DH + off);
            cp_async16(&vt[buf][row][off], Vp + (size_t)row * SEQ + j * 64 + off);
        }
    };

    // Q chunk: 4 async per thread; overlap with first K/V tile.
    #pragma unroll
    for (int i = 0; i < 4; ++i) {
        int c = tid + 256 * i;
        int row = c >> 3, off = (c & 7) * 8;
        cp_async16(&qs[row][off], Qp + (size_t)(qt * 128 + row) * DH + off);
    }
    issue_kv(0, 0);
    cp_async_wait<4>();        // drain Q's 4 oldest; kv tile 0 may stay in flight
    __syncthreads();

    v16bf qa[2];
    qa[0] = load_fragA(&qs[w * 16 + (lane & 15)][0], 0, g);
    qa[1] = load_fragA(&qs[w * 16 + (lane & 15)][0], 32, g);

    v8f O[4];
    float mrow[8], lrow[8];
    #pragma unroll
    for (int nt = 0; nt < 4; ++nt)
        #pragma unroll
        for (int r = 0; r < 8; ++r) O[nt][r] = 0.0f;
    #pragma unroll
    for (int r = 0; r < 8; ++r) { mrow[r] = -1.0e30f; lrow[r] = 0.0f; }

    constexpr int NJ = SEQ / 64;
    for (int j = 0; j < NJ; ++j) {
        const int cur = j & 1;
        if (j + 1 < NJ) {
            issue_kv(j + 1, cur ^ 1);   // overlap next tile's DMA with compute
            cp_async_wait<4>();
        } else {
            cp_async_wait<0>();
        }
        __syncthreads();

        // S = Q * K^T * scale
        v8f st[4];
        #pragma unroll
        for (int nt = 0; nt < 4; ++nt) {
            v8f a;
            #pragma unroll
            for (int r = 0; r < 8; ++r) a[r] = 0.0f;
            #pragma unroll
            for (int kk = 0; kk < 2; ++kk) {
                v16bf bfrag = load_fragB(&ks[cur][nt * 16 + (lane & 15)][0], kk * 32, g);
                a = wmma_bf16(qa[kk], bfrag, a);
            }
            #pragma unroll
            for (int r = 0; r < 8; ++r) st[nt][r] = a[r] * SCORE_SCALE;
        }

        // online softmax (row r+8g lives on one 16-lane half)
        float mnew[8], rscale[8];
        #pragma unroll
        for (int r = 0; r < 8; ++r) {
            float x = fmaxf(fmaxf(st[0][r], st[1][r]), fmaxf(st[2][r], st[3][r]));
            x = fmaxf(x, __shfl_xor(x, 1, 32));
            x = fmaxf(x, __shfl_xor(x, 2, 32));
            x = fmaxf(x, __shfl_xor(x, 4, 32));
            x = fmaxf(x, __shfl_xor(x, 8, 32));
            float mn = fmaxf(mrow[r], x);
            rscale[r] = __expf(mrow[r] - mn);
            mnew[r] = mn;
        }
        #pragma unroll
        for (int nt = 0; nt < 4; ++nt)
            #pragma unroll
            for (int r = 0; r < 8; ++r) {
                O[nt][r] *= rscale[r];
                st[nt][r] = __expf(st[nt][r] - mnew[r]);
            }
        #pragma unroll
        for (int r = 0; r < 8; ++r) {
            float s = st[0][r] + st[1][r] + st[2][r] + st[3][r];
            s += __shfl_xor(s, 1, 32);
            s += __shfl_xor(s, 2, 32);
            s += __shfl_xor(s, 4, 32);
            s += __shfl_xor(s, 8, 32);
            lrow[r] = lrow[r] * rscale[r] + s;
            mrow[r] = mnew[r];
        }

        // P: C-layout f32 -> LDS bf16 -> A-layout fragments
        #pragma unroll
        for (int nt = 0; nt < 4; ++nt)
            #pragma unroll
            for (int r = 0; r < 8; ++r)
                ps[w][8 * g + r][nt * 16 + (lane & 15)] =
                    (unsigned short)bf16_bits(st[nt][r]);
        asm volatile("s_wait_dscnt 0" ::: "memory");

        // O += P * V
        #pragma unroll
        for (int kk = 0; kk < 2; ++kk) {
            v16bf pa = load_fragA(&ps[w][lane & 15][0], kk * 32, g);
            #pragma unroll
            for (int nt = 0; nt < 4; ++nt) {
                v16bf bfrag = load_fragB(&vt[cur][nt * 16 + (lane & 15)][0], kk * 32, g);
                O[nt] = wmma_bf16(pa, bfrag, O[nt]);
            }
        }
        __syncthreads();             // protect buffer before it is re-filled
    }

    // epilogue: normalize, store f32 [B,S,D_OUT]
    #pragma unroll
    for (int nt = 0; nt < 4; ++nt) {
        int d = nt * 16 + (lane & 15);
        #pragma unroll
        for (int r = 0; r < 8; ++r) {
            int srow = qt * 128 + w * 16 + 8 * g + r;
            out[((size_t)b_ * SEQ + srow) * D_OUT + h * DH + d] = O[nt][r] / lrow[r];
        }
    }
}

// ---------------------------------------------------------------------------
// Launch
// ---------------------------------------------------------------------------
extern "C" void kernel_launch(void* const* d_in, const int* in_sizes, int n_in,
                              void* d_out, int out_size, void* d_ws, size_t ws_size,
                              hipStream_t stream) {
    (void)in_sizes; (void)n_in; (void)out_size; (void)ws_size;
    const float* X  = (const float*)d_in[0];
    const float* Wq = (const float*)d_in[1];
    const float* bq = (const float*)d_in[2];
    const float* Wk = (const float*)d_in[3];
    const float* bk = (const float*)d_in[4];
    const float* Wv = (const float*)d_in[5];
    const float* bv = (const float*)d_in[6];
    float* out = (float*)d_out;

    // workspace layout (bf16/ushort elements)
    const size_t XB_N  = (size_t)BATCH * SEQ * D_IN;    // 4,194,304
    const size_t WT_N  = (size_t)3 * D_IN * D_OUT;      // 3,145,728
    const size_t QKV_N = (size_t)BATCH * NH * SEQ * DH; // 4,194,304
    unsigned short* xb  = (unsigned short*)d_ws;
    unsigned short* wtb = xb + XB_N;
    unsigned short* qws = wtb + WT_N;
    unsigned short* kws = qws + QKV_N;
    unsigned short* vws = kws + QKV_N;

    cvt_x_kernel<<<dim3((unsigned)(XB_N / (256 * 8))), 256, 0, stream>>>(X, xb);
    cvt_wT_kernel<<<dim3(16, 16, 3), 256, 0, stream>>>(Wq, Wk, Wv, wtb);

    dim3 g1(BATCH * SEQ / 128, D_OUT / 64, 3);
    qkv_proj_kernel<<<g1, 256, 0, stream>>>(xb, wtb, bq, bk, bv, qws, kws, vws);

    dim3 g2(BATCH * NH * (SEQ / 128));
    attn_kernel<<<g2, 256, 0, stream>>>(qws, kws, vws, out);
}
